// InteractionPPBlock_5952824672720
// MI455X (gfx1250) — matrix operand order repaired and checked by reading, not compiled, not run on hardware
//
#include <hip/hip_runtime.h>

#define HID  128
#define INTD 64
#define NRADD 6

typedef __attribute__((ext_vector_type(16))) __bf16 bf16x16;
typedef __attribute__((ext_vector_type(8)))  __bf16 bf16x8;
typedef __attribute__((ext_vector_type(8)))  float  f32x8;

// ---------- helpers ----------
__device__ __forceinline__ float silu(float v) {
  // v * sigmoid(v) using single-instruction v_rcp_f32 (avoids IEEE div expansion)
  return v * __builtin_amdgcn_rcpf(1.0f + __expf(-v));
}

__device__ __forceinline__ f32x8 wmma_bf16(bf16x16 a, bf16x16 b, f32x8 c) {
  return __builtin_amdgcn_wmma_f32_16x16x32_bf16(false, a, false, b, (short)0, c, false, false);
}

__device__ __forceinline__ bf16x16 cat8(bf16x8 lo, bf16x8 hi) {
  return __builtin_shufflevector(lo, hi, 0, 1, 2, 3, 4, 5, 6, 7, 8, 9, 10, 11, 12, 13, 14, 15);
}

__device__ __forceinline__ bf16x8 zero8() {
  bf16x8 z;
#pragma unroll
  for (int e = 0; e < 8; ++e) z[e] = (__bf16)0.0f;
  return z;
}

// A fragment (16x32 bf16, M=lane&15) from fp32 row-major global; K runs are 32B-aligned
__device__ __forceinline__ bf16x16 afrag_g(const float* rowptr, int kc, int half, int Klimit) {
  const int c0 = kc * 32 + half * 8;        // elements 0..7
  const int c1 = kc * 32 + 16 + half * 8;   // elements 8..15
  bf16x8 lo = zero8(), hi = zero8();
  if (c0 < Klimit) lo = __builtin_convertvector(*(const f32x8*)(rowptr + c0), bf16x8);
  if (c1 < Klimit) hi = __builtin_convertvector(*(const f32x8*)(rowptr + c1), bf16x8);
  return cat8(lo, hi);
}

// A fragment from LDS bf16 tile [16][HID]
__device__ __forceinline__ bf16x16 afrag_l(const __bf16* tile, int kc, int lane) {
  const int m = lane & 15, half = lane >> 4;
  const int c0 = kc * 32 + half * 8;
  const int c1 = kc * 32 + 16 + half * 8;
  bf16x8 lo = *(const bf16x8*)(tile + m * HID + c0);
  bf16x8 hi = *(const bf16x8*)(tile + m * HID + c1);
  return cat8(lo, hi);
}

// B fragment from pre-packed weights (32B contiguous per lane)
__device__ __forceinline__ bf16x16 bfrag(const __bf16* wp, int frag, int lane) {
  return *(const bf16x16*)(wp + ((size_t)frag * 32 + (size_t)lane) * 16);
}

// ---------- prep kernels ----------
__global__ void zero_f(float* __restrict__ p, long n) {
  long i = (long)blockIdx.x * blockDim.x + threadIdx.x;
  if (i < n) p[i] = 0.f;
}

__global__ void combine_rbf(const float* __restrict__ w1, const float* __restrict__ w2,
                            float* __restrict__ dst) { // [6][128]
  int i = blockIdx.x * blockDim.x + threadIdx.x;
  if (i >= NRADD * HID) return;
  int r = i / HID, c = i % HID;
  float s = 0.f;
  for (int m = 0; m < 8; ++m) s += w1[r * 8 + m] * w2[m * HID + c];
  dst[i] = s;
}

__global__ void combine_sbf(const float* __restrict__ w1, const float* __restrict__ w2,
                            float* __restrict__ dst) { // [56][64]
  int i = blockIdx.x * blockDim.x + threadIdx.x;
  if (i >= 56 * 64) return;
  int r = i >> 6, c = i & 63;
  float s = 0.f;
  for (int m = 0; m < 8; ++m) s += w1[r * 8 + m] * w2[m * 64 + c];
  dst[i] = s;
}

// pack fp32 [Ksrc x N] row-major weight into WMMA-B fragment-major bf16; grid=(Kpad/32)*(N/16), block=32
__global__ void pack_b(const float* __restrict__ src, __bf16* __restrict__ dst,
                       int Ksrc, int N) {
  const int lane = threadIdx.x & 31;
  const int frag = blockIdx.x;
  const int nT = N >> 4;
  const int kc = frag / nT, nc = frag % nT;
  const int col = nc * 16 + (lane & 15);
  const int kb  = kc * 32 + (lane >> 4) * 8;
  __bf16* o = dst + ((size_t)frag * 32 + lane) * 16;
#pragma unroll
  for (int e = 0; e < 16; ++e) {
    int K = kb + (e < 8 ? e : 8 + e);
    float v = (K < Ksrc) ? src[(size_t)K * N + col] : 0.f;
    o[e] = (__bf16)v;
  }
}

// ---------- stage 1: per-edge GEMMs ----------
__global__ __launch_bounds__(256) void edge_in_kernel(
    const float* __restrict__ x, const float* __restrict__ rbf, const float* __restrict__ wrbfc,
    const __bf16* __restrict__ Wji, const __bf16* __restrict__ Wkj,
    const __bf16* __restrict__ Wdn,
    const float* __restrict__ bji, const float* __restrict__ bkj,
    float* __restrict__ xji_out, float* __restrict__ xdn_out, int E) {
  __shared__ __align__(16) __bf16 tbuf[8][16 * HID];
  __shared__ float rbfs[8][16 * 8];
  const int lane = threadIdx.x & 31, w = threadIdx.x >> 5;
  const int m = lane & 15, half = lane >> 4;
  const long row0 = ((long)blockIdx.x * 8 + w) * 16;
  long arow = row0 + m; if (arow >= E) arow = E - 1;
  if (lane < 16) {
    long rr = row0 + lane; if (rr >= E) rr = E - 1;
#pragma unroll
    for (int r = 0; r < NRADD; ++r) rbfs[w][lane * 8 + r] = rbf[rr * NRADD + r];
  }
  const float* xrow = x + arow * HID;
  bf16x16 xa[4];
#pragma unroll
  for (int kc = 0; kc < 4; ++kc) xa[kc] = afrag_g(xrow, kc, half, HID);
  asm volatile("s_wait_dscnt 0" ::: "memory");
#pragma unroll
  for (int nc = 0; nc < 8; ++nc) {
    const int col = nc * 16 + m;
    f32x8 a1, a2;
    const float b1 = bji[col], b2 = bkj[col];
#pragma unroll
    for (int j = 0; j < 8; ++j) { a1[j] = b1; a2[j] = b2; }
#pragma unroll
    for (int kc = 0; kc < 4; ++kc) {
      a1 = wmma_bf16(xa[kc], bfrag(Wji, kc * 8 + nc, lane), a1);
      a2 = wmma_bf16(xa[kc], bfrag(Wkj, kc * 8 + nc, lane), a2);
    }
    float wr[NRADD];
#pragma unroll
    for (int r = 0; r < NRADD; ++r) wr[r] = wrbfc[r * HID + col];
#pragma unroll
    for (int j = 0; j < 8; ++j) {
      const int lr = j + half * 8;
      long orow = row0 + lr;
      float v1 = silu(a1[j]);
      if (orow < E) xji_out[orow * HID + col] = v1;
      float re = 0.f;
#pragma unroll
      for (int r = 0; r < NRADD; ++r) re += rbfs[w][lr * 8 + r] * wr[r];
      tbuf[w][lr * HID + col] = (__bf16)(silu(a2[j]) * re);
    }
  }
  asm volatile("s_wait_dscnt 0" ::: "memory");
  bf16x16 ta[4];
#pragma unroll
  for (int kc = 0; kc < 4; ++kc) ta[kc] = afrag_l(tbuf[w], kc, lane);
#pragma unroll
  for (int nc = 0; nc < 4; ++nc) {
    f32x8 acc;
#pragma unroll
    for (int j = 0; j < 8; ++j) acc[j] = 0.f;
#pragma unroll
    for (int kc = 0; kc < 4; ++kc) acc = wmma_bf16(ta[kc], bfrag(Wdn, kc * 4 + nc, lane), acc);
    const int col = nc * 16 + m;
#pragma unroll
    for (int j = 0; j < 8; ++j) {
      long orow = row0 + j + half * 8;
      if (orow < E) xdn_out[orow * INTD + col] = silu(acc[j]);
    }
  }
}

// ---------- stage 2: sbf projection + gather + scatter-add ----------
__global__ __launch_bounds__(256) void triplet_kernel(
    const float* __restrict__ sbf, const __bf16* __restrict__ Wsbf,
    const float* __restrict__ xdn, const int* __restrict__ idx_kj, const int* __restrict__ idx_ji,
    float* __restrict__ agg, long T) {
  const int lane = threadIdx.x & 31, w = threadIdx.x >> 5;
  const long row0 = ((long)blockIdx.x * 8 + w) * 16;
  if (row0 >= T) return;                       // wave-uniform, no barriers in this kernel
  const int m = lane & 15, half = lane >> 4;
  long arow = row0 + m; if (arow >= T) arow = T - 1;
  const float* srow = sbf + arow * 56;
  bf16x16 sa[2];
#pragma unroll
  for (int kc = 0; kc < 2; ++kc) sa[kc] = afrag_g(srow, kc, half, 56);
  int gk[8], gj[8]; bool okr[8];
#pragma unroll
  for (int j = 0; j < 8; ++j) {
    long r = row0 + j + half * 8;
    okr[j] = (r < T);
    long rc = okr[j] ? r : 0;
    gk[j] = idx_kj[rc]; gj[j] = idx_ji[rc];
  }
#pragma unroll
  for (int nc = 0; nc < 4; ++nc) {
    f32x8 acc;
#pragma unroll
    for (int j = 0; j < 8; ++j) acc[j] = 0.f;
#pragma unroll
    for (int kc = 0; kc < 2; ++kc) acc = wmma_bf16(sa[kc], bfrag(Wsbf, kc * 4 + nc, lane), acc);
    const int col = nc * 16 + m;
#pragma unroll
    for (int j = 0; j < 8; ++j) {
      if (okr[j]) {
        float v = acc[j] * xdn[(long)gk[j] * INTD + col];
        atomicAdd(&agg[(long)gj[j] * INTD + col], v);
      }
    }
  }
}

// ---------- stage 3: W_up / W_lin / W_out + residuals + BN stats ----------
__global__ __launch_bounds__(256) void edge_out_kernel(
    const float* __restrict__ agg, const float* __restrict__ xji, const float* __restrict__ x,
    const __bf16* __restrict__ Wup, const __bf16* __restrict__ Wlin,
    const __bf16* __restrict__ Wout,
    const float* __restrict__ blin, const float* __restrict__ bout,
    float* __restrict__ out, float* __restrict__ gsum, float* __restrict__ gsq, int E) {
  __shared__ __align__(16) __bf16 tbuf[8][16 * HID];
  __shared__ float csum[HID], csq[HID];
  if (threadIdx.x < HID) { csum[threadIdx.x] = 0.f; csq[threadIdx.x] = 0.f; }
  __syncthreads();
  const int lane = threadIdx.x & 31, w = threadIdx.x >> 5;
  const int m = lane & 15, half = lane >> 4;
  const long row0 = ((long)blockIdx.x * 8 + w) * 16;
  long arow = row0 + m; if (arow >= E) arow = E - 1;
  bf16x16 ga[2];
#pragma unroll
  for (int kc = 0; kc < 2; ++kc) ga[kc] = afrag_g(agg + arow * INTD, kc, half, INTD);
#pragma unroll
  for (int nc = 0; nc < 8; ++nc) {
    f32x8 acc;
#pragma unroll
    for (int j = 0; j < 8; ++j) acc[j] = 0.f;
#pragma unroll
    for (int kc = 0; kc < 2; ++kc) acc = wmma_bf16(ga[kc], bfrag(Wup, kc * 8 + nc, lane), acc);
    const int col = nc * 16 + m;
#pragma unroll
    for (int j = 0; j < 8; ++j) {
      const int lr = j + half * 8;
      long orow = row0 + lr;
      long rr = (orow < E) ? orow : (E - 1);
      float h = xji[rr * HID + col] + silu(acc[j]);
      tbuf[w][lr * HID + col] = (__bf16)h;
    }
  }
  asm volatile("s_wait_dscnt 0" ::: "memory");
  bf16x16 ha[4];
#pragma unroll
  for (int kc = 0; kc < 4; ++kc) ha[kc] = afrag_l(tbuf[w], kc, lane);
#pragma unroll
  for (int nc = 0; nc < 8; ++nc) {
    const int col = nc * 16 + m;
    f32x8 acc;
    const float bl = blin[col];
#pragma unroll
    for (int j = 0; j < 8; ++j) acc[j] = bl;
#pragma unroll
    for (int kc = 0; kc < 4; ++kc) acc = wmma_bf16(ha[kc], bfrag(Wlin, kc * 8 + nc, lane), acc);
#pragma unroll
    for (int j = 0; j < 8; ++j) {
      const int lr = j + half * 8;
      long orow = row0 + lr;
      long rr = (orow < E) ? orow : (E - 1);
      float h2 = silu(acc[j]) + x[rr * HID + col];
      tbuf[w][lr * HID + col] = (__bf16)h2;
    }
  }
  asm volatile("s_wait_dscnt 0" ::: "memory");
#pragma unroll
  for (int kc = 0; kc < 4; ++kc) ha[kc] = afrag_l(tbuf[w], kc, lane);
#pragma unroll
  for (int nc = 0; nc < 8; ++nc) {
    const int col = nc * 16 + m;
    f32x8 acc;
    const float bo = bout[col];
#pragma unroll
    for (int j = 0; j < 8; ++j) acc[j] = bo;
#pragma unroll
    for (int kc = 0; kc < 4; ++kc) acc = wmma_bf16(ha[kc], bfrag(Wout, kc * 8 + nc, lane), acc);
    float s = 0.f, q = 0.f;
#pragma unroll
    for (int j = 0; j < 8; ++j) {
      long orow = row0 + j + half * 8;
      if (orow < E) {
        float h3 = silu(acc[j]);
        out[orow * HID + col] = h3;
        s += h3; q += h3 * h3;
      }
    }
    atomicAdd(&csum[col], s);
    atomicAdd(&csq[col], q);
  }
  __syncthreads();
  if (threadIdx.x < HID) {
    atomicAdd(&gsum[threadIdx.x], csum[threadIdx.x]);
    atomicAdd(&gsq[threadIdx.x], csq[threadIdx.x]);
  }
}

// ---------- BN finalize + apply ----------
__global__ void bn_stats(const float* __restrict__ gsum, const float* __restrict__ gsq,
                         const float* __restrict__ gamma, const float* __restrict__ beta,
                         float* __restrict__ scale, float* __restrict__ shift, float invE) {
  int c = threadIdx.x;
  float mean = gsum[c] * invE;
  float var  = gsq[c] * invE - mean * mean;
  float sc   = gamma[c] * rsqrtf(var + 1e-5f);
  scale[c] = sc;
  shift[c] = beta[c] - mean * sc;
}

__global__ void bn_apply(float* __restrict__ out, const float* __restrict__ scale,
                         const float* __restrict__ shift, long n) {
  long i = ((long)blockIdx.x * blockDim.x + threadIdx.x) * 4;
  if (i >= n) return;
  float4 v = *(float4*)(out + i);
  int c = (int)(i & (HID - 1));
  float4 s = *(const float4*)(scale + c);
  float4 t = *(const float4*)(shift + c);
  v.x = v.x * s.x + t.x; v.y = v.y * s.y + t.y;
  v.z = v.z * s.z + t.z; v.w = v.w * s.w + t.w;
  *(float4*)(out + i) = v;
}

// ---------- host ----------
extern "C" void kernel_launch(void* const* d_in, const int* in_sizes, int n_in,
                              void* d_out, int out_size, void* d_ws, size_t ws_size,
                              hipStream_t stream) {
  const float* x     = (const float*)d_in[0];
  const float* rbf   = (const float*)d_in[2];
  const float* sbf   = (const float*)d_in[3];
  const int*   ikj   = (const int*)d_in[4];
  const int*   iji   = (const int*)d_in[5];
  const float* Wrbf1 = (const float*)d_in[6];
  const float* Wrbf2 = (const float*)d_in[7];
  const float* Wsbf1 = (const float*)d_in[8];
  const float* Wsbf2 = (const float*)d_in[9];
  const float* Wkj   = (const float*)d_in[10];
  const float* bkj   = (const float*)d_in[11];
  const float* Wji   = (const float*)d_in[12];
  const float* bji   = (const float*)d_in[13];
  const float* Wdn   = (const float*)d_in[14];
  const float* Wup   = (const float*)d_in[15];
  const float* Wlin  = (const float*)d_in[16];
  const float* blin  = (const float*)d_in[17];
  const float* Wout  = (const float*)d_in[18];
  const float* bout  = (const float*)d_in[19];
  const float* gamma = (const float*)d_in[20];
  const float* beta  = (const float*)d_in[21];
  (void)n_in; (void)out_size; (void)ws_size;

  const long E = in_sizes[0] / HID;
  const long T = in_sizes[4];

  char* wsb = (char*)d_ws;
  size_t off = 0;
  auto alloc = [&](size_t bytes) -> char* {
    char* p = wsb + off;
    off = (off + bytes + 255) & ~(size_t)255;
    return p;
  };
  float* xji   = (float*)alloc((size_t)E * HID * 4);
  float* xdn   = (float*)alloc((size_t)E * INTD * 4);
  float* agg   = (float*)alloc((size_t)E * INTD * 4);
  float* stats = (float*)alloc(512 * 4);           // gsum | gsq | scale | shift
  float* gsum  = stats, *gsq = stats + 128, *scale = stats + 256, *shift = stats + 384;
  float* wrbfc = (float*)alloc(NRADD * HID * 4);
  float* wsbfc = (float*)alloc(56 * 64 * 4);
  __bf16* Wji_p  = (__bf16*)alloc(128 * 128 * 2);
  __bf16* Wkj_p  = (__bf16*)alloc(128 * 128 * 2);
  __bf16* Wdn_p  = (__bf16*)alloc(128 * 64 * 2);
  __bf16* Wup_p  = (__bf16*)alloc(64 * 128 * 2);
  __bf16* Wlin_p = (__bf16*)alloc(128 * 128 * 2);
  __bf16* Wout_p = (__bf16*)alloc(128 * 128 * 2);
  __bf16* Wsbf_p = (__bf16*)alloc(64 * 64 * 2);

  // zero agg + BN accumulators
  long nz = E * INTD;
  zero_f<<<(unsigned)((nz + 255) / 256), 256, 0, stream>>>(agg, nz);
  zero_f<<<1, 256, 0, stream>>>(gsum, 256);

  // fold small chains, pack weights for WMMA-B layout
  combine_rbf<<<(NRADD * HID + 255) / 256, 256, 0, stream>>>(Wrbf1, Wrbf2, wrbfc);
  combine_sbf<<<(56 * 64 + 255) / 256, 256, 0, stream>>>(Wsbf1, Wsbf2, wsbfc);
  pack_b<<<4 * 8, 32, 0, stream>>>(Wji,   Wji_p, 128, 128);
  pack_b<<<4 * 8, 32, 0, stream>>>(Wkj,   Wkj_p, 128, 128);
  pack_b<<<4 * 4, 32, 0, stream>>>(Wdn,   Wdn_p, 128, 64);
  pack_b<<<2 * 8, 32, 0, stream>>>(Wup,   Wup_p, 64, 128);
  pack_b<<<4 * 8, 32, 0, stream>>>(Wlin,  Wlin_p, 128, 128);
  pack_b<<<4 * 8, 32, 0, stream>>>(Wout,  Wout_p, 128, 128);
  pack_b<<<2 * 4, 32, 0, stream>>>(wsbfc, Wsbf_p, 56, 64);

  // stage 1
  unsigned eblocks = (unsigned)(((E + 15) / 16 + 7) / 8);
  edge_in_kernel<<<eblocks, 256, 0, stream>>>(x, rbf, wrbfc, Wji_p, Wkj_p, Wdn_p,
                                              bji, bkj, xji, xdn, (int)E);
  // stage 2
  unsigned tblocks = (unsigned)(((T + 15) / 16 + 7) / 8);
  triplet_kernel<<<tblocks, 256, 0, stream>>>(sbf, Wsbf_p, xdn, ikj, iji, agg, T);
  // stage 3
  edge_out_kernel<<<eblocks, 256, 0, stream>>>(agg, xji, x, Wup_p, Wlin_p, Wout_p,
                                               blin, bout, (float*)d_out, gsum, gsq, (int)E);
  // BN
  bn_stats<<<1, 128, 0, stream>>>(gsum, gsq, gamma, beta, scale, shift, 1.0f / (float)E);
  long nv = (E * HID) / 4;
  bn_apply<<<(unsigned)((nv + 255) / 256), 256, 0, stream>>>((float*)d_out, scale, shift, E * HID);
}
